// GCN_29257317220561
// MI455X (gfx1250) — compile-verified, matrix operand order checked by reading, not run on hardware
//
#include <hip/hip_runtime.h>

typedef __bf16 bf16;
typedef __attribute__((ext_vector_type(16))) __bf16 v16bf;
typedef __attribute__((ext_vector_type(8)))  __bf16 v8bf;
typedef __attribute__((ext_vector_type(8)))  float  v8f;

// ---------------- utility kernels ----------------

__global__ void zero_f4(float4* __restrict__ p, long long n4) {
    long long i = (long long)blockIdx.x * blockDim.x + threadIdx.x;
    if (i < n4) p[i] = make_float4(0.f, 0.f, 0.f, 0.f);
}

__global__ void degree_accum(const int* __restrict__ src, const int* __restrict__ dst,
                             float* __restrict__ deg_s, float* __restrict__ deg_d, int e) {
    int i = blockIdx.x * blockDim.x + threadIdx.x;
    if (i < e) {
        atomicAdd(&deg_s[src[i]], 1.0f);
        atomicAdd(&deg_d[dst[i]], 1.0f);
    }
}

__global__ void rsqrt_clamp(float* __restrict__ v, int n) {
    int i = blockIdx.x * blockDim.x + threadIdx.x;
    if (i < n) v[i] = rsqrtf(fmaxf(v[i], 1.0f));
}

// ---------------- SpMM scatter: agg[dst] += h[src] * rs[src] ----------------
// one thread per (edge, 4-feature chunk); for d=128 one wave handles one edge.

__global__ void scatter_add(const float* __restrict__ h, const int* __restrict__ src,
                            const int* __restrict__ dst, const float* __restrict__ rs,
                            float* __restrict__ agg, int n_edges, int d) {
    const int chunks = d >> 2;
    long long gid = (long long)blockIdx.x * blockDim.x + threadIdx.x;
    if (gid >= (long long)n_edges * chunks) return;
    int e = (int)(gid / chunks);
    int c = (int)(gid - (long long)e * chunks);
    int s = src[e], t = dst[e];
    float sc = rs[s];
    float4 v = *(const float4*)(h + (size_t)s * d + c * 4);
    float* o = agg + (size_t)t * d + c * 4;
    atomicAdd(o + 0, v.x * sc);
    atomicAdd(o + 1, v.y * sc);
    atomicAdd(o + 2, v.z * sc);
    atomicAdd(o + 3, v.w * sc);
}

// ---------------- WMMA GEMM: out = relu(rd[i] * (agg @ W) + b) ----------------
// block: one 16-row tile of agg; wave w: 16-col tile [16w, 16w+16).
// Weights transposed into LDS as bf16: sWT[n * d_in + k].

__global__ void gemm_bias_relu_wmma(const float* __restrict__ agg,
                                    const float* __restrict__ W,
                                    const float* __restrict__ bias,
                                    const float* __restrict__ rd,
                                    float* __restrict__ out,
                                    int n_rows, int d_in, int d_out) {
    extern __shared__ bf16 sWT[];
    const int tid = threadIdx.x;
    for (int idx = tid; idx < d_in * d_out; idx += blockDim.x) {
        int k = idx / d_out;
        int n = idx - k * d_out;
        sWT[n * d_in + k] = (bf16)W[idx];
    }
    __syncthreads();

    const int wave = tid >> 5;
    const int lane = tid & 31;
    const int m    = lane & 15;
    const int koff = (lane >> 4) << 3;            // 0 or 8 (K-half select)
    const int col  = wave * 16 + m;               // this lane's B/N column
    const int row  = blockIdx.x * 16 + m;         // this lane's A row
    const int arow = (row < n_rows) ? row : (n_rows - 1);

    const float* aRow = agg + (size_t)arow * d_in;
    const bf16*  wCol = sWT + col * d_in;

    v8f acc = {};
    for (int k0 = 0; k0 < d_in; k0 += 32) {
        // A fragment: K runs [k0+koff, +8) and [k0+koff+16, +8), fp32 -> bf16
        float4 a0 = *(const float4*)(aRow + k0 + koff);
        float4 a1 = *(const float4*)(aRow + k0 + koff + 4);
        float4 a2 = *(const float4*)(aRow + k0 + koff + 16);
        float4 a3 = *(const float4*)(aRow + k0 + koff + 20);
        v16bf a;
        a[0]=(bf16)a0.x;  a[1]=(bf16)a0.y;  a[2]=(bf16)a0.z;  a[3]=(bf16)a0.w;
        a[4]=(bf16)a1.x;  a[5]=(bf16)a1.y;  a[6]=(bf16)a1.z;  a[7]=(bf16)a1.w;
        a[8]=(bf16)a2.x;  a[9]=(bf16)a2.y;  a[10]=(bf16)a2.z; a[11]=(bf16)a2.w;
        a[12]=(bf16)a3.x; a[13]=(bf16)a3.y; a[14]=(bf16)a3.z; a[15]=(bf16)a3.w;

        // B fragment: same K-run layout, contiguous bf16 in LDS (two 16B loads)
        v8bf b0 = *(const v8bf*)(wCol + k0 + koff);
        v8bf b1 = *(const v8bf*)(wCol + k0 + koff + 16);
        v16bf b = __builtin_shufflevector(b0, b1, 0,1,2,3,4,5,6,7,8,9,10,11,12,13,14,15);

        acc = __builtin_amdgcn_wmma_f32_16x16x32_bf16(
            /*neg_a=*/false, a, /*neg_b=*/false, b,
            /*c_mod=*/(short)0, acc, /*reuse_a=*/false, /*reuse_b=*/false);
    }

    const float bcol = bias[col];
    const int mhi = (lane >> 4) << 3;  // C/D layout: lanes 16-31 hold M = r+8
    #pragma unroll
    for (int r = 0; r < 8; ++r) {
        int orow = blockIdx.x * 16 + r + mhi;
        if (orow < n_rows) {
            float v = acc[r] * rd[orow] + bcol;
            out[(size_t)orow * d_out + col] = fmaxf(v, 0.0f);
        }
    }
}

// ---------------- host launcher ----------------

extern "C" void kernel_launch(void* const* d_in, const int* in_sizes, int n_in,
                              void* d_out, int out_size, void* d_ws, size_t ws_size,
                              hipStream_t stream) {
    const float* x  = (const float*)d_in[0];
    const int*  src = (const int*) d_in[1];
    const int*  dst = (const int*) d_in[2];
    const float* W1 = (const float*)d_in[3];
    const float* b1 = (const float*)d_in[4];
    const float* W2 = (const float*)d_in[5];
    const float* b2 = (const float*)d_in[6];
    const float* W3 = (const float*)d_in[7];
    const float* b3 = (const float*)d_in[8];

    const int E    = in_sizes[1];
    const int D_H  = in_sizes[4];           // 128
    const int D_IN = in_sizes[3] / D_H;     // 128
    const int D_O  = in_sizes[8];           // 64
    const int N    = in_sizes[0] / D_IN;    // 50000

    float* rs  = (float*)d_ws;                 // N: rsqrt(deg_src)
    float* rd  = rs + N;                       // N: rsqrt(deg_dst)
    float* agg = rd + N;                       // N * max(d_in)
    float* h   = agg + (size_t)N * D_IN;       // N * D_H hidden activations

    const int B = 256;
    auto cdiv = [](long long a, long long b) { return (int)((a + b - 1) / b); };

    // degrees (shared by all layers)
    long long degf4 = (2LL * N) / 4;
    zero_f4<<<cdiv(degf4, B), B, 0, stream>>>((float4*)rs, degf4);
    degree_accum<<<cdiv(E, B), B, 0, stream>>>(src, dst, rs, rd, E);
    rsqrt_clamp<<<cdiv(2 * N, B), B, 0, stream>>>(rs, 2 * N);

    auto layer = [&](const float* hin, int din, const float* W, const float* bias,
                     float* hout, int dout) {
        long long aggf4 = (long long)N * din / 4;
        zero_f4<<<cdiv(aggf4, B), B, 0, stream>>>((float4*)agg, aggf4);
        long long total = (long long)E * (din / 4);
        scatter_add<<<cdiv(total, B), B, 0, stream>>>(hin, src, dst, rs, agg, E, din);
        int waves = dout / 16;
        size_t shmem = (size_t)din * dout * sizeof(bf16);
        gemm_bias_relu_wmma<<<cdiv(N, 16), 32 * waves, shmem, stream>>>(
            agg, W, bias, rd, hout, N, din, dout);
    };

    layer(x, D_IN, W1, b1, h, D_H);          // layer 1: 128 -> 128
    layer(h, D_H,  W2, b2, h, D_H);          // layer 2: 128 -> 128 (in-place safe: stream order)
    layer(h, D_H,  W3, b3, (float*)d_out, D_O); // layer 3: 128 -> 64 into d_out
}
